// MultiHeadAttention_38233798869772
// MI455X (gfx1250) — compile-verified, matrix operand order checked by reading, not run on hardware
//
#include <hip/hip_runtime.h>

// ---------------- CDNA5 (gfx1250) multi-head attention, wave32 WMMA + TDM ---------

typedef _Float16 h16;
typedef __attribute__((ext_vector_type(16))) _Float16 v16h;
typedef __attribute__((ext_vector_type(8)))  _Float16 v8h;
typedef __attribute__((ext_vector_type(8)))  float    v8f;
typedef __attribute__((ext_vector_type(4)))  unsigned int v4u;
typedef __attribute__((ext_vector_type(8)))  int      v8i;
typedef __attribute__((ext_vector_type(4)))  int      v4i;

#define HEADS  16
#define DK     64
#define DMODEL 1024
#define SEQ    2048
#define BATCH  4
#define LPITCH 72   // LDS tile pitch in halves: 64 + 8 (TDM pad: 4 dwords / 32 dwords)
#define BUFSTRIDE (2 * 64 * LPITCH)   // halves per double-buffer slot (K tile + V tile)

// Tensor Data Mover builtin: 5-arg on ROCm clang-22, 6-arg on clang-23 toolchain.
#if defined(__has_builtin)
#if __has_builtin(__builtin_amdgcn_tensor_load_to_lds) && \
    __has_builtin(__builtin_amdgcn_s_wait_tensorcnt)
#define USE_TDM 1
#endif
#endif
#ifndef USE_TDM
#define USE_TDM 0
#endif

// D = A(16x32 f16) * B(32x16 f16) + C(16x16 f32)
__device__ __forceinline__ v8f wmma16(v16h a, v16h b, v8f c) {
  return __builtin_amdgcn_wmma_f32_16x16x32_f16(false, a, false, b, (short)0, c,
                                                false, false);
}

// 16x32 f16 A-style fragment (also used for B^T) from row-major f16 memory
// (global or LDS).  ISA 7.12.2: per lane two contiguous 8-half chunks at
// k0=(lane>>4)*8 and k0+16.
__device__ __forceinline__ v16h frag_h(const h16* __restrict__ t, int ld) {
  int l = threadIdx.x & 31;
  const h16* p = t + (long)(l & 15) * ld + ((l >> 4) << 3);
  v8h lo = *(const v8h*)(p);
  v8h hi = *(const v8h*)(p + 16);
  return __builtin_shufflevector(lo, hi, 0, 1, 2, 3, 4, 5, 6, 7,
                                 8, 9, 10, 11, 12, 13, 14, 15);
}

#if USE_TDM
// TDM 2-D tile load: 64 rows x 64 halves -> LDS with pitch 144B.
// D# group0: count=1, lds_addr, global_addr, type=2.  group1: data_size=2B,
// pad_enable, pad_interval=32 dwords (code 4), pad_amount=4 dwords (code 3),
// tensor_dim0/1 = tile_dim0/1 = 64, dim0 stride in elements.
__device__ __forceinline__ void tdm_load_2d(const h16* gsrc, h16* ldst,
                                            unsigned int stride_elems) {
  unsigned long long ga = (unsigned long long)(__SIZE_TYPE__)gsrc;
  unsigned int la = (unsigned int)(__SIZE_TYPE__)ldst;  // low 32b = LDS offset
  v4u g0;
  g0[0] = 1u;                                           // count=1 (user mode)
  g0[1] = la;
  g0[2] = (unsigned int)(ga & 0xffffffffu);
  g0[3] = (unsigned int)((ga >> 32) & 0x1ffffffu) | (2u << 30);  // type=2
  v8i g1;
  g1[0] = (int)((1u << 16) | (1u << 20) | (4u << 22) | (3u << 25));
  g1[1] = (int)(64u << 16);        // tensor_dim0[15:0] in bits[31:16]
  g1[2] = (int)(64u << 16);        // tensor_dim0[31:16]=0 | tensor_dim1[15:0]
  g1[3] = (int)(64u << 16);        // tensor_dim1[31:16]=0 | tile_dim0=64
  g1[4] = (int)(64u);              // tile_dim1=64 | tile_dim2=0
  g1[5] = (int)stride_elems;       // tensor_dim0_stride[31:0]
  g1[6] = 0;
  g1[7] = 0;
  v4i z = {0, 0, 0, 0};
#if __clang_major__ >= 23
  v8i zz = {0, 0, 0, 0, 0, 0, 0, 0};
  __builtin_amdgcn_tensor_load_to_lds(g0, g1, z, z, zz, 0);
#else
  __builtin_amdgcn_tensor_load_to_lds(g0, g1, z, z, 0);
#endif
}
#endif

// Fallback cooperative stage: 128 threads copy two 64x64 f16 tiles (b128 each).
__device__ __forceinline__ void stage_copy(const h16* __restrict__ Kg,
                                           const h16* __restrict__ Vg,
                                           h16* lK, h16* lV) {
  int tid = threadIdx.x;
#pragma unroll
  for (int t = 0; t < 4; ++t) {
    int c = tid + t * 128;               // 0..511 chunks of 8 halves
    int row = c >> 3, co = (c & 7) << 3;
    *(v8h*)(lK + row * LPITCH + co) = *(const v8h*)(Kg + (long)row * DK + co);
    *(v8h*)(lV + row * LPITCH + co) = *(const v8h*)(Vg + (long)row * SEQ + co);
  }
}

// ---------------- elementwise f32 -> f16 (one-time precision convert) -----------
__global__ void cvt_f16(const float* __restrict__ src, h16* __restrict__ dst,
                        long n) {
  long i = ((long)blockIdx.x * blockDim.x + threadIdx.x) * 8;
  if (i >= n) return;
  float4 a = *(const float4*)(src + i);
  float4 b = *(const float4*)(src + i + 4);
  v8h o;
  o[0] = (h16)a.x; o[1] = (h16)a.y; o[2] = (h16)a.z; o[3] = (h16)a.w;
  o[4] = (h16)b.x; o[5] = (h16)b.y; o[6] = (h16)b.z; o[7] = (h16)b.w;
  *(v8h*)(dst + i) = o;
}

// ---------------- Y = X @ W^T + b : one wave computes 32(M) x 64(N) ------------
// MODE 0: f16 head-split [B,H,S,DK]; MODE 1: f16 [B,H,DK,S]; MODE 2: f32 out.
template <int MODE>
__global__ void gemm_proj(const h16* __restrict__ X, const h16* __restrict__ Wh,
                          const float* __restrict__ bias, void* __restrict__ Yv) {
  const int MT = (BATCH * SEQ) / 32;
  int wave = blockIdx.x * (blockDim.x >> 5) + (threadIdx.x >> 5);
  int mt = wave % MT;
  int hs = wave / MT;
  int m0 = mt * 32;
  int l  = threadIdx.x & 31;
  int col = l & 15, rbase = (l >> 4) << 3;

  v8f acc[2][4] = {};
  for (int kk = 0; kk < DMODEL; kk += 32) {
    __builtin_prefetch(X + (long)m0 * DMODEL + kk + 64, 0, 3);
    v16h a0 = frag_h(X + (long)m0 * DMODEL + kk, DMODEL);
    v16h a1 = frag_h(X + (long)(m0 + 16) * DMODEL + kk, DMODEL);
#pragma unroll
    for (int t = 0; t < 4; ++t) {
      v16h b = frag_h(Wh + (long)(hs * 64 + t * 16) * DMODEL + kk, DMODEL);
      acc[0][t] = wmma16(a0, b, acc[0][t]);
      acc[1][t] = wmma16(a1, b, acc[1][t]);
    }
  }

  int bidx = m0 / SEQ;
#pragma unroll
  for (int g = 0; g < 2; ++g) {
    int s0 = (m0 % SEQ) + g * 16 + rbase;
#pragma unroll
    for (int t = 0; t < 4; ++t) {
      float bi = bias[hs * 64 + t * 16 + col];
      if (MODE == 1) {
        v8h pk;
#pragma unroll
        for (int r = 0; r < 8; ++r) pk[r] = (h16)(acc[g][t][r] + bi);
        *(v8h*)((h16*)Yv +
                (((long)(bidx * HEADS + hs) * DK + t * 16 + col) * SEQ + s0)) = pk;
      } else {
#pragma unroll
        for (int r = 0; r < 8; ++r) {
          float vv = acc[g][t][r] + bi;
          if (MODE == 0) {
            ((h16*)Yv)[(((long)(bidx * HEADS + hs) * SEQ + s0 + r) * DK) +
                       t * 16 + col] = (h16)vv;
          } else {
            ((float*)Yv)[((long)bidx * SEQ + s0 + r) * DMODEL + hs * 64 +
                         t * 16 + col] = vv;
          }
        }
      }
    }
  }
}

// ---------------- flash attention, transposed scores, TDM-staged K/V ------------
// Block = 4 waves, one (b,h), 128 queries; wave = 32 query columns (2 tiles).
// K/V 64-key tiles double-buffered in LDS via TENSOR_LOAD_TO_LDS.
__global__ void __launch_bounds__(128)
flash_attn(const h16* __restrict__ Qh, const h16* __restrict__ Kh,
           const h16* __restrict__ Vt, h16* __restrict__ X2) {
  __shared__ h16 lds[2 * BUFSTRIDE];                     // 2 bufs x (K+V) 36 KB
  int w = threadIdx.x >> 5;
  int l = threadIdx.x & 31;
  int bh   = blockIdx.x >> 4;                            // 16 blocks per (b,h)
  int qblk = blockIdx.x & 15;
  int b = bh / HEADS, h = bh % HEADS;
  int q0 = qblk * 128 + w * 32;
  int col = l & 15, rbase = (l >> 4) << 3;

  const h16* Qp = Qh + ((long)bh * SEQ + q0) * DK;
  const h16* Kp = Kh + (long)bh * SEQ * DK;
  const h16* Vp = Vt + (long)bh * DK * SEQ;
  h16* lbase = &lds[0];                                  // single addrspacecast

  v16h bq[4];                                            // B-frags of (Q/8)^T
  bq[0] = frag_h(Qp, DK);
  bq[1] = frag_h(Qp + 32, DK);
  bq[2] = frag_h(Qp + 16 * DK, DK);
  bq[3] = frag_h(Qp + 16 * DK + 32, DK);
#pragma unroll
  for (int f = 0; f < 4; ++f)
#pragma unroll
    for (int i = 0; i < 16; ++i) bq[f][i] *= (h16)0.125f;

  v8f o[2][4] = {};                                      // O^T per q-tile
  float runmax[2] = {-1e30f, -1e30f}, runsum[2] = {0.f, 0.f};
  const int NB = SEQ / 64;

#if USE_TDM
  if (w == 0) {
    tdm_load_2d(Kp, lbase, DK);
    tdm_load_2d(Vp, lbase + 64 * LPITCH, SEQ);
    __builtin_amdgcn_s_wait_tensorcnt(0);
  }
#else
  stage_copy(Kp, Vp, lbase, lbase + 64 * LPITCH);
#endif
  __syncthreads();

  for (int j = 0; j < NB; ++j) {
    int cur = j & 1, nxt = cur ^ 1;
    if (j + 1 < NB) {                                    // overlap DMA w/ math
      long kb1 = (long)(j + 1) * 64;
      h16* nK = lbase + nxt * BUFSTRIDE;
#if USE_TDM
      if (w == 0) {
        tdm_load_2d(Kp + kb1 * DK, nK, DK);
        tdm_load_2d(Vp + kb1, nK + 64 * LPITCH, SEQ);
      }
#else
      stage_copy(Kp + kb1 * DK, Vp + kb1, nK, nK + 64 * LPITCH);
#endif
    }
    const h16* lK = lbase + cur * BUFSTRIDE;
    const h16* lV = lK + 64 * LPITCH;
#pragma unroll
    for (int qs = 0; qs < 2; ++qs) {
      // ---- S^T block: 64(keys) x 16(q) from LDS K tile ---------------------
      v8f sc[4];
#pragma unroll
      for (int t = 0; t < 4; ++t) {
        v8f z = {};
        z     = wmma16(frag_h(lK + t * 16 * LPITCH, LPITCH), bq[2 * qs], z);
        sc[t] = wmma16(frag_h(lK + t * 16 * LPITCH + 32, LPITCH), bq[2 * qs + 1], z);
      }
      // ---- online softmax: per-lane over 32 regs + one shfl_xor(16) --------
      float bm = sc[0][0];
#pragma unroll
      for (int t = 0; t < 4; ++t)
#pragma unroll
        for (int i = 0; i < 8; ++i) bm = fmaxf(bm, sc[t][i]);
      bm = fmaxf(bm, __shfl_xor(bm, 16, 32));
      float nm    = fmaxf(runmax[qs], bm);
      float alpha = __expf(runmax[qs] - nm);
      runmax[qs]  = nm;
      float rs = 0.f;
      v16h pb0, pb1;                                     // P^T B-frags, direct
#pragma unroll
      for (int i = 0; i < 8; ++i) {
        float p0 = __expf(sc[0][i] - nm); rs += p0; pb0[i]     = (h16)p0;
        float p1 = __expf(sc[1][i] - nm); rs += p1; pb0[8 + i] = (h16)p1;
        float p2 = __expf(sc[2][i] - nm); rs += p2; pb1[i]     = (h16)p2;
        float p3 = __expf(sc[3][i] - nm); rs += p3; pb1[8 + i] = (h16)p3;
      }
      rs = rs + __shfl_xor(rs, 16, 32);
      runsum[qs] = runsum[qs] * alpha + rs;
      // ---- O^T = alpha*O^T + V^T P^T from LDS V tile -----------------------
#pragma unroll
      for (int u = 0; u < 4; ++u) {
#pragma unroll
        for (int r = 0; r < 8; ++r) o[qs][u][r] *= alpha;
        v8f z = wmma16(frag_h(lV + u * 16 * LPITCH, LPITCH), pb0, o[qs][u]);
        o[qs][u] = wmma16(frag_h(lV + u * 16 * LPITCH + 32, LPITCH), pb1, z);
      }
    }
#if USE_TDM
    if (w == 0) __builtin_amdgcn_s_wait_tensorcnt(0);
#endif
    __syncthreads();
  }

  // ---- normalize, write f16 [B*S, DMODEL]: 8 consecutive d per lane -> b128
#pragma unroll
  for (int qs = 0; qs < 2; ++qs) {
    float inv = 1.f / runsum[qs];
    long row = (long)b * SEQ + q0 + qs * 16 + col;
#pragma unroll
    for (int u = 0; u < 4; ++u) {
      v8h pk;
#pragma unroll
      for (int r = 0; r < 8; ++r) pk[r] = (h16)(o[qs][u][r] * inv);
      *(v8h*)(X2 + row * DMODEL + h * 64 + u * 16 + rbase) = pk;
    }
  }
}

extern "C" void kernel_launch(void* const* d_in, const int* in_sizes, int n_in,
                              void* d_out, int out_size, void* d_ws, size_t ws_size,
                              hipStream_t stream) {
  const float* q  = (const float*)d_in[0];
  const float* k  = (const float*)d_in[1];
  const float* v  = (const float*)d_in[2];
  const float* wq = (const float*)d_in[3];
  const float* bq = (const float*)d_in[4];
  const float* wk = (const float*)d_in[5];
  const float* bk = (const float*)d_in[6];
  const float* wv = (const float*)d_in[7];
  const float* bv = (const float*)d_in[8];
  const float* wo = (const float*)d_in[9];
  const float* bo = (const float*)d_in[10];

  char* ws = (char*)d_ws;                          // 120 MB used
  h16* Qh  = (h16*)(ws + ((size_t)0 << 20));
  h16* Kh  = (h16*)(ws + ((size_t)16 << 20));
  h16* Vt  = (h16*)(ws + ((size_t)32 << 20));
  h16* X2  = (h16*)(ws + ((size_t)48 << 20));
  h16* Xq  = (h16*)(ws + ((size_t)64 << 20));
  h16* Xk  = (h16*)(ws + ((size_t)80 << 20));
  h16* Xv  = (h16*)(ws + ((size_t)96 << 20));
  h16* Wqh = (h16*)(ws + ((size_t)112 << 20));
  h16* Wkh = (h16*)(ws + ((size_t)114 << 20));
  h16* Wvh = (h16*)(ws + ((size_t)116 << 20));
  h16* Woh = (h16*)(ws + ((size_t)118 << 20));

  const long NX = (long)BATCH * SEQ * DMODEL;
  const long NW = (long)DMODEL * DMODEL;
  cvt_f16<<<dim3((NX / 8 + 255) / 256), dim3(256), 0, stream>>>(q, Xq, NX);
  cvt_f16<<<dim3((NX / 8 + 255) / 256), dim3(256), 0, stream>>>(k, Xk, NX);
  cvt_f16<<<dim3((NX / 8 + 255) / 256), dim3(256), 0, stream>>>(v, Xv, NX);
  cvt_f16<<<dim3((NW / 8 + 255) / 256), dim3(256), 0, stream>>>(wq, Wqh, NW);
  cvt_f16<<<dim3((NW / 8 + 255) / 256), dim3(256), 0, stream>>>(wk, Wkh, NW);
  cvt_f16<<<dim3((NW / 8 + 255) / 256), dim3(256), 0, stream>>>(wv, Wvh, NW);
  cvt_f16<<<dim3((NW / 8 + 255) / 256), dim3(256), 0, stream>>>(wo, Woh, NW);

  const int gjobs = ((BATCH * SEQ) / 32) * HEADS;  // 4096 waves
  dim3 blk(128);
  gemm_proj<0><<<dim3(gjobs / 4), blk, 0, stream>>>(Xq, Wqh, bq, Qh);
  gemm_proj<0><<<dim3(gjobs / 4), blk, 0, stream>>>(Xk, Wkh, bk, Kh);
  gemm_proj<1><<<dim3(gjobs / 4), blk, 0, stream>>>(Xv, Wvh, bv, Vt);

  // 1024 blocks x 128 threads: block = 128 queries of one (b,h)
  flash_attn<<<dim3((BATCH * HEADS) * (SEQ / 128)), blk, 0, stream>>>(Qh, Kh, Vt,
                                                                      X2);

  gemm_proj<2><<<dim3(gjobs / 4), blk, 0, stream>>>(X2, Woh, bo, (float*)d_out);
}
